// Graph2Cone128_49572512530721
// MI455X (gfx1250) — matrix operand order repaired and checked by reading, not compiled, not run on hardware
//
#include <hip/hip_runtime.h>
#include <hip/hip_bf16.h>
#include <math.h>

// ---------------- problem constants (match reference) ----------------
#define NC 128          // channels
#define NN 50000        // nodes
#define NE 600000       // edges
#define NG 256          // graphs
#define BN_EPS 1e-5f
#define PI_F 3.14159265358979323846f

// ---------------- WMMA types (CDNA5 / gfx1250, wave32) ----------------
typedef __attribute__((ext_vector_type(16))) __bf16 v16bf;
typedef __attribute__((ext_vector_type(8)))  float  v8f;

union Frag { v16bf v; uint4 q[2]; };

// =====================================================================
// GEMM: Y[M x 128] = act( X[M x 128] @ W[128 x 128] + bias )
// Wt is bf16, stored N-major: Wt[n*128 + k] = W[k][n]
// Block: 128 threads = 4 waves; each block computes 64 rows x 128 cols.
// Each wave: one 16-row strip, 8 N-tiles, K loop of 4 x v_wmma bf16.
// TAIL=false blocks are fully in-range (branch-free epilogue);
// the (at most one) ragged block is launched separately with TAIL=true.
// bias must always be a valid pointer (zero vector if no bias).
// =====================================================================
template<int ACT, bool TAIL>  // ACT: 0 = none, 1 = tanh
__global__ __launch_bounds__(128)
void gemm128_wmma(const float* __restrict__ X, const __bf16* __restrict__ Wt,
                  const float* __restrict__ bias, float* __restrict__ Y,
                  int rowStart, int M)
{
    __shared__ __align__(16) __bf16 sA[64][136];   // stride 272B -> 16B aligned rows
    __shared__ __align__(16) __bf16 sB[128][136];  // [n][k]

    const int tid     = threadIdx.x;
    const int rowBase = rowStart + blockIdx.x * 64;

    // prefetch next block's A tile into L2 (gfx1250 global_prefetch_b8)
    if (!TAIL && tid == 0)
        __builtin_prefetch(X + (size_t)(rowBase + 64) * NC, 0, 1);

    // stage B (already bf16, [n][k])
    for (int i = tid; i < 128 * 128; i += 128) {
        int n = i >> 7, k = i & 127;
        sB[n][k] = Wt[i];
    }
    // stage A with f32 -> bf16 conversion
    if (!TAIL) {
        for (int i = tid; i < 64 * 128; i += 128) {
            int r = i >> 7, k = i & 127;
            sA[r][k] = (__bf16)X[(size_t)(rowBase + r) * NC + k];
        }
    } else {
        for (int i = tid; i < 64 * 128; i += 128) {
            int r = i >> 7, k = i & 127;
            int gr = rowBase + r;
            float v = (gr < M) ? X[(size_t)gr * NC + k] : 0.0f;
            sA[r][k] = (__bf16)v;
        }
    }
    __syncthreads();

    const int wave = tid >> 5;
    const int lane = tid & 31;
    const int rl   = lane & 15;   // row (A) / col (B,C) within 16
    const int g    = lane >> 4;   // lane group
    const int aRow = wave * 16 + rl;

    // A fragments for all 4 K-steps, held in VGPRs across the N loop
    Frag A0, A1, A2, A3;
    {
        const __bf16* pa = &sA[aRow][8 * g];
        A0.q[0] = *(const uint4*)(pa);       A0.q[1] = *(const uint4*)(pa + 16);
        A1.q[0] = *(const uint4*)(pa + 32);  A1.q[1] = *(const uint4*)(pa + 48);
        A2.q[0] = *(const uint4*)(pa + 64);  A2.q[1] = *(const uint4*)(pa + 80);
        A3.q[0] = *(const uint4*)(pa + 96);  A3.q[1] = *(const uint4*)(pa + 112);
    }

    for (int nt = 0; nt < 8; ++nt) {
        v8f acc = {};
        const __bf16* pb = &sB[nt * 16 + rl][8 * g];
        Frag B;
        B.q[0] = *(const uint4*)(pb);       B.q[1] = *(const uint4*)(pb + 16);
        acc = __builtin_amdgcn_wmma_f32_16x16x32_bf16(false, A0.v, false, B.v, (short)0, acc, false, false);
        B.q[0] = *(const uint4*)(pb + 32);  B.q[1] = *(const uint4*)(pb + 48);
        acc = __builtin_amdgcn_wmma_f32_16x16x32_bf16(false, A1.v, false, B.v, (short)0, acc, false, false);
        B.q[0] = *(const uint4*)(pb + 64);  B.q[1] = *(const uint4*)(pb + 80);
        acc = __builtin_amdgcn_wmma_f32_16x16x32_bf16(false, A2.v, false, B.v, (short)0, acc, false, false);
        B.q[0] = *(const uint4*)(pb + 96);  B.q[1] = *(const uint4*)(pb + 112);
        acc = __builtin_amdgcn_wmma_f32_16x16x32_bf16(false, A3.v, false, B.v, (short)0, acc, false, false);

        const int col = nt * 16 + rl;
        const float b = bias[col];
        if (!TAIL) {
            // branch-free epilogue: one s_clause of 8 global_store_b32
            float* yp = Y + (size_t)(rowBase + wave * 16 + 8 * g) * NC + col;
            #pragma unroll
            for (int r = 0; r < 8; ++r) {
                float v = acc[r] + b;
                if (ACT == 1) v = tanhf(v);
                yp[(size_t)r * NC] = v;
            }
        } else {
            #pragma unroll
            for (int r = 0; r < 8; ++r) {
                int grow = rowBase + wave * 16 + r + 8 * g;
                if (grow < M) {
                    float v = acc[r] + b;
                    if (ACT == 1) v = tanhf(v);
                    Y[(size_t)grow * NC + col] = v;
                }
            }
        }
    }
}

// host-side launcher: full-tile grid + optional ragged tail block
template<int ACT>
static void launch_gemm128(const float* X, const __bf16* Wt, const float* bias,
                           float* Y, int M, hipStream_t stream)
{
    const int full = M / 64;
    if (full > 0)
        gemm128_wmma<ACT, false><<<full, 128, 0, stream>>>(X, Wt, bias, Y, 0, M);
    if (M % 64)
        gemm128_wmma<ACT, true><<<1, 128, 0, stream>>>(X, Wt, bias, Y, full * 64, M);
}

// =====================================================================
// Elementwise / scatter kernels
// =====================================================================
__device__ __forceinline__ void atomicMaxF(float* addr, float val)
{
    unsigned int* ua = (unsigned int*)addr;
    unsigned int old = __float_as_uint(*addr);
    while (__uint_as_float(old) < val) {
        unsigned int assumed = old;
        old = atomicCAS(ua, assumed, __float_as_uint(val));
        if (old == assumed) break;
    }
}

__global__ void fill_kernel(float* p, float v, int n)
{
    int i = blockIdx.x * blockDim.x + threadIdx.x;
    if (i < n) p[i] = v;
}

// bf16 weight transpose: Wt[n][k] = W[k][n]
__global__ void wconv_kernel(const float* __restrict__ W, __bf16* __restrict__ Wt)
{
    int i = blockIdx.x * blockDim.x + threadIdx.x;   // 16384
    if (i >= 128 * 128) return;
    int k = i >> 7, n = i & 127;
    Wt[(size_t)n * 128 + k] = (__bf16)W[i];
}

// deg[dst] += 1 ; sumW[dst] += eaWe[e]   (wave per edge)
__global__ void edge_accum_kernel(const int* __restrict__ dst,
                                  const float* __restrict__ eaWe,
                                  float* __restrict__ deg, float* __restrict__ sumW)
{
    int gid  = blockIdx.x * blockDim.x + threadIdx.x;
    int e    = gid >> 5;
    int lane = gid & 31;
    if (e >= NE) return;
    int d = dst[e];
    if (lane == 0) atomicAdd(&deg[d], 1.0f);
    const float* row = eaWe + (size_t)e * NC;
    float* out = sumW + (size_t)d * NC;
    for (int c = lane; c < NC; c += 32) atomicAdd(&out[c], row[c]);
}

// a[e] = sum_c leaky(xl[s]+xr[d]+eaWe[e]) * att ; amax[d] = max(...)
__global__ void edge_a_kernel(const int* __restrict__ src, const int* __restrict__ dst,
                              const float* __restrict__ xl, const float* __restrict__ xr,
                              const float* __restrict__ eaWe, const float* __restrict__ att,
                              float* __restrict__ a, float* __restrict__ amax)
{
    int gid  = blockIdx.x * blockDim.x + threadIdx.x;
    int e    = gid >> 5;
    int lane = gid & 31;
    if (e >= NE) return;
    int s = src[e], d = dst[e];
    float part = 0.0f;
    for (int c = lane; c < NC; c += 32) {
        float m = xl[(size_t)s * NC + c] + xr[(size_t)d * NC + c] + eaWe[(size_t)e * NC + c];
        float lr = (m > 0.0f) ? m : 0.2f * m;
        part += lr * att[c];
    }
    for (int off = 16; off > 0; off >>= 1) part += __shfl_down(part, off, 32);
    if (lane == 0) { a[e] = part; atomicMaxF(&amax[d], part); }
}

// self-loop score: m = xl[i]+xr[i]+sumW[i]/max(deg,1)
__global__ void node_self_a_kernel(const float* __restrict__ xl, const float* __restrict__ xr,
                                   const float* __restrict__ sumW, const float* __restrict__ deg,
                                   const float* __restrict__ att,
                                   float* __restrict__ aself, float* __restrict__ amax)
{
    int gid  = blockIdx.x * blockDim.x + threadIdx.x;
    int i    = gid >> 5;
    int lane = gid & 31;
    if (i >= NN) return;
    float invd = 1.0f / fmaxf(deg[i], 1.0f);
    float part = 0.0f;
    for (int c = lane; c < NC; c += 32) {
        float m = xl[(size_t)i * NC + c] + xr[(size_t)i * NC + c] + sumW[(size_t)i * NC + c] * invd;
        float lr = (m > 0.0f) ? m : 0.2f * m;
        part += lr * att[c];
    }
    for (int off = 16; off > 0; off >>= 1) part += __shfl_down(part, off, 32);
    if (lane == 0) { aself[i] = part; atomicMaxF(&amax[i], part); }
}

__global__ void edge_exp_kernel(const int* __restrict__ dst, const float* __restrict__ a,
                                const float* __restrict__ amax,
                                float* __restrict__ ex, float* __restrict__ den)
{
    int e = blockIdx.x * blockDim.x + threadIdx.x;
    if (e >= NE) return;
    int d = dst[e];
    float v = __expf(a[e] - amax[d]);
    ex[e] = v;
    atomicAdd(&den[d], v);
}

__global__ void node_exp_kernel(const float* __restrict__ aself, const float* __restrict__ amax,
                                float* __restrict__ exs, float* __restrict__ den)
{
    int i = blockIdx.x * blockDim.x + threadIdx.x;
    if (i >= NN) return;
    float v = __expf(aself[i] - amax[i]);
    exs[i] = v;
    atomicAdd(&den[i], v);
}

// h[i] = bo + alpha_self * xl[i]   (single-writer init before edge scatter)
__global__ void node_out_init_kernel(const float* __restrict__ xl, const float* __restrict__ exs,
                                     const float* __restrict__ den, const float* __restrict__ bo,
                                     float* __restrict__ h)
{
    int idx = blockIdx.x * blockDim.x + threadIdx.x;
    if (idx >= NN * NC) return;
    int i = idx >> 7, c = idx & 127;
    float d = den[i];
    float alpha = (d > 0.0f) ? exs[i] / d : 0.0f;
    h[idx] = bo[c] + alpha * xl[idx];
}

// h[dst] += (ex/den[dst]) * xl[src]   (wave per edge)
__global__ void edge_scatter_kernel(const int* __restrict__ src, const int* __restrict__ dst,
                                    const float* __restrict__ xl, const float* __restrict__ ex,
                                    const float* __restrict__ den, float* __restrict__ h)
{
    int gid  = blockIdx.x * blockDim.x + threadIdx.x;
    int e    = gid >> 5;
    int lane = gid & 31;
    if (e >= NE) return;
    int s = src[e], d = dst[e];
    float dn = den[d];
    float alpha = (dn > 0.0f) ? ex[e] / dn : 0.0f;
    const float* xs = xl + (size_t)s * NC;
    float* hd = h + (size_t)d * NC;
    for (int c = lane; c < NC; c += 32) atomicAdd(&hd[c], alpha * xs[c]);
}

// ---------------- BatchNorm ----------------
__global__ __launch_bounds__(128)
void bn_stats_kernel(const float* __restrict__ h, float* __restrict__ sum, float* __restrict__ sq)
{
    int c = threadIdx.x;            // 128 threads
    float s = 0.0f, s2 = 0.0f;
    for (int i = blockIdx.x; i < NN; i += gridDim.x) {
        float v = h[(size_t)i * NC + c];
        s += v; s2 += v * v;
    }
    atomicAdd(&sum[c], s);
    atomicAdd(&sq[c], s2);
}

__global__ void bn_apply_tanh_kernel(float* __restrict__ h, const float* __restrict__ sum,
                                     const float* __restrict__ sq, const float* __restrict__ gam,
                                     const float* __restrict__ bet)
{
    int idx = blockIdx.x * blockDim.x + threadIdx.x;
    if (idx >= NN * NC) return;
    int c = idx & 127;
    float mu  = sum[c] * (1.0f / NN);
    float var = sq[c] * (1.0f / NN) - mu * mu;
    float v = (h[idx] - mu) * rsqrtf(var + BN_EPS) * gam[c] + bet[c];
    h[idx] = tanhf(v);
}

// ---------------- attentional pooling ----------------
__global__ void pool_max_kernel(const int* __restrict__ batch, const float* __restrict__ gate,
                                float* __restrict__ gmax)
{
    int idx = blockIdx.x * blockDim.x + threadIdx.x;
    if (idx >= NN * NC) return;
    int i = idx >> 7, c = idx & 127;
    atomicMaxF(&gmax[(size_t)batch[i] * NC + c], gate[idx]);
}

__global__ void pool_den_kernel(const int* __restrict__ batch, float* __restrict__ gate,
                                const float* __restrict__ gmax, float* __restrict__ gden)
{
    int idx = blockIdx.x * blockDim.x + threadIdx.x;
    if (idx >= NN * NC) return;
    int i = idx >> 7, c = idx & 127;
    float eg = __expf(gate[idx] - gmax[(size_t)batch[i] * NC + c]);
    gate[idx] = eg;                                  // recomputed each call -> deterministic
    atomicAdd(&gden[(size_t)batch[i] * NC + c], eg);
}

__global__ void pool_sum_kernel(const int* __restrict__ batch, const float* __restrict__ eg,
                                const float* __restrict__ gden, const float* __restrict__ h,
                                float* __restrict__ pooled)
{
    int idx = blockIdx.x * blockDim.x + threadIdx.x;
    if (idx >= NN * NC) return;
    int i = idx >> 7, c = idx & 127;
    size_t gi = (size_t)batch[i] * NC + c;
    float d = gden[gi];
    float w = (d > 0.0f) ? eg[idx] / d : 0.0f;
    atomicAdd(&pooled[gi], w * h[idx]);
}

// split: axis = o[:, :64]*PI ; aperture = (o[:, 64:]+1)*PI ; concat flat
__global__ void split_out_kernel(const float* __restrict__ o, float* __restrict__ out)
{
    int idx = blockIdx.x * blockDim.x + threadIdx.x;
    if (idx >= NG * NC) return;
    int g = idx >> 7, c = idx & 127;
    float v = o[idx];
    if (c < 64) out[g * 64 + c] = v * PI_F;
    else        out[NG * 64 + g * 64 + (c - 64)] = (v + 1.0f) * PI_F;
}

// =====================================================================
// Host orchestration
// =====================================================================
static void run_gat_layer(const float* xin, const int* src, const int* dst,
                          const __bf16* WlT, const float* bl,
                          const __bf16* WrT, const float* br,
                          const __bf16* WeT, const float* att, const float* bo,
                          const float* edge_attr, const float* zerobias,
                          float* xl, float* xr, float* eaWe, float* deg, float* sumW,
                          float* a, float* aself, float* amax, float* den,
                          float* ex, float* exs, float* hout, hipStream_t stream)
{
    const int WPE = (NE * 32 + 255) / 256;   // wave-per-edge blocks
    const int WPN = (NN * 32 + 255) / 256;
    const int FN  = (NN + 255) / 256;
    const int FE  = (NE + 255) / 256;
    const int FNC = (NN * NC + 255) / 256;

    launch_gemm128<0>(xin, WlT, bl, xl, NN, stream);
    launch_gemm128<0>(xin, WrT, br, xr, NN, stream);
    launch_gemm128<0>(edge_attr, WeT, zerobias, eaWe, NE, stream);

    fill_kernel<<<FN, 256, 0, stream>>>(deg, 0.0f, NN);
    fill_kernel<<<FNC, 256, 0, stream>>>(sumW, 0.0f, NN * NC);
    fill_kernel<<<FN, 256, 0, stream>>>(amax, -3.0e38f, NN);
    fill_kernel<<<FN, 256, 0, stream>>>(den, 0.0f, NN);

    edge_accum_kernel<<<WPE, 256, 0, stream>>>(dst, eaWe, deg, sumW);
    edge_a_kernel<<<WPE, 256, 0, stream>>>(src, dst, xl, xr, eaWe, att, a, amax);
    node_self_a_kernel<<<WPN, 256, 0, stream>>>(xl, xr, sumW, deg, att, aself, amax);
    edge_exp_kernel<<<FE, 256, 0, stream>>>(dst, a, amax, ex, den);
    node_exp_kernel<<<FN, 256, 0, stream>>>(aself, amax, exs, den);
    node_out_init_kernel<<<FNC, 256, 0, stream>>>(xl, exs, den, bo, hout);
    edge_scatter_kernel<<<WPE, 256, 0, stream>>>(src, dst, xl, ex, den, hout);
}

extern "C" void kernel_launch(void* const* d_in, const int* in_sizes, int n_in,
                              void* d_out, int out_size, void* d_ws, size_t ws_size,
                              hipStream_t stream)
{
    (void)in_sizes; (void)n_in; (void)out_size; (void)ws_size;

    // ---- inputs (setup_inputs order) ----
    const float* x    = (const float*)d_in[0];
    const int*   eidx = (const int*)d_in[1];
    const float* ea   = (const float*)d_in[2];
    const int*   batch= (const int*)d_in[3];
    const float* Wl1  = (const float*)d_in[4];  const float* bl1 = (const float*)d_in[5];
    const float* Wr1  = (const float*)d_in[6];  const float* br1 = (const float*)d_in[7];
    const float* We1  = (const float*)d_in[8];  const float* att1= (const float*)d_in[9];
    const float* bo1  = (const float*)d_in[10];
    const float* Wl2  = (const float*)d_in[11]; const float* bl2 = (const float*)d_in[12];
    const float* Wr2  = (const float*)d_in[13]; const float* br2 = (const float*)d_in[14];
    const float* We2  = (const float*)d_in[15]; const float* att2= (const float*)d_in[16];
    const float* bo2  = (const float*)d_in[17];
    const float* g1   = (const float*)d_in[18]; const float* be1 = (const float*)d_in[19];
    const float* g2   = (const float*)d_in[20]; const float* be2 = (const float*)d_in[21];
    const float* Ag1  = (const float*)d_in[22]; const float* bg1 = (const float*)d_in[23];
    const float* Ag2  = (const float*)d_in[24]; const float* bg2 = (const float*)d_in[25];
    const float* Wf   = (const float*)d_in[26]; const float* bf  = (const float*)d_in[27];

    const int* src = eidx;
    const int* dst = eidx + NE;

    // ---- workspace carve-up ----
    float* w = (float*)d_ws;
    size_t o = 0;
    auto alloc = [&](size_t n) { float* p = w + o; o += n; return p; };
    float* xl    = alloc((size_t)NN * NC);
    float* xr    = alloc((size_t)NN * NC);
    float* sumW  = alloc((size_t)NN * NC);
    float* h1    = alloc((size_t)NN * NC);
    float* h2    = alloc((size_t)NN * NC);
    float* eaWe  = alloc((size_t)NE * NC);
    float* deg   = alloc(NN);
    float* a     = alloc(NE);
    float* aself = alloc(NN);
    float* amax  = alloc(NN);
    float* den   = alloc(NN);
    float* ex    = alloc(NE);
    float* exs   = alloc(NN);
    float* gmax  = alloc((size_t)NG * NC);
    float* gden  = alloc((size_t)NG * NC);
    float* pooled= alloc((size_t)NG * NC);
    float* obuf  = alloc((size_t)NG * NC);
    float* bnsum = alloc(NC);
    float* bnsq  = alloc(NC);
    float* zbias = alloc(NC);          // zero bias vector for bias-less GEMMs
    __bf16* wbf  = (__bf16*)(w + o);   // 9 x 128x128 bf16 transposed weights

    fill_kernel<<<1, 256, 0, stream>>>(zbias, 0.0f, NC);

    // ---- convert weights to bf16 (N-major) ----
    const float* wsrc[9] = { Wl1, Wr1, We1, Wl2, Wr2, We2, Ag1, Ag2, Wf };
    for (int i = 0; i < 9; ++i)
        wconv_kernel<<<64, 256, 0, stream>>>(wsrc[i], wbf + (size_t)i * 128 * 128);
    const __bf16* WlT1 = wbf;               const __bf16* WrT1 = wbf + 16384;
    const __bf16* WeT1 = wbf + 2 * 16384;
    const __bf16* WlT2 = wbf + 3 * 16384;   const __bf16* WrT2 = wbf + 4 * 16384;
    const __bf16* WeT2 = wbf + 5 * 16384;
    const __bf16* Ag1T = wbf + 6 * 16384;   const __bf16* Ag2T = wbf + 7 * 16384;
    const __bf16* WfT  = wbf + 8 * 16384;

    const int FNC = (NN * NC + 255) / 256;
    const int FGC = (NG * NC + 255) / 256;

    // ---- layer 1: GATv2 -> BN -> tanh ----
    run_gat_layer(x, src, dst, WlT1, bl1, WrT1, br1, WeT1, att1, bo1, ea, zbias,
                  xl, xr, eaWe, deg, sumW, a, aself, amax, den, ex, exs, h1, stream);
    fill_kernel<<<1, 256, 0, stream>>>(bnsum, 0.0f, NC);
    fill_kernel<<<1, 256, 0, stream>>>(bnsq, 0.0f, NC);
    bn_stats_kernel<<<512, 128, 0, stream>>>(h1, bnsum, bnsq);
    bn_apply_tanh_kernel<<<FNC, 256, 0, stream>>>(h1, bnsum, bnsq, g1, be1);

    // ---- layer 2: GATv2 -> BN -> tanh ----
    run_gat_layer(h1, src, dst, WlT2, bl2, WrT2, br2, WeT2, att2, bo2, ea, zbias,
                  xl, xr, eaWe, deg, sumW, a, aself, amax, den, ex, exs, h2, stream);
    fill_kernel<<<1, 256, 0, stream>>>(bnsum, 0.0f, NC);
    fill_kernel<<<1, 256, 0, stream>>>(bnsq, 0.0f, NC);
    bn_stats_kernel<<<512, 128, 0, stream>>>(h2, bnsum, bnsq);
    bn_apply_tanh_kernel<<<FNC, 256, 0, stream>>>(h2, bnsum, bnsq, g2, be2);

    // ---- attentional pooling (gate MLP via WMMA, reuse xl/xr buffers) ----
    float* gate1 = xl;   // tanh(h2 @ Ag1 + bg1)
    float* gate  = xr;   // gate1 @ Ag2 + bg2
    launch_gemm128<1>(h2, Ag1T, bg1, gate1, NN, stream);
    launch_gemm128<0>(gate1, Ag2T, bg2, gate, NN, stream);

    fill_kernel<<<FGC, 256, 0, stream>>>(gmax, -3.0e38f, NG * NC);
    fill_kernel<<<FGC, 256, 0, stream>>>(gden, 0.0f, NG * NC);
    fill_kernel<<<FGC, 256, 0, stream>>>(pooled, 0.0f, NG * NC);
    pool_max_kernel<<<FNC, 256, 0, stream>>>(batch, gate, gmax);
    pool_den_kernel<<<FNC, 256, 0, stream>>>(batch, gate, gmax, gden);
    pool_sum_kernel<<<FNC, 256, 0, stream>>>(batch, gate, gden, h2, pooled);

    // ---- final linear + tanh + split ----
    launch_gemm128<1>(pooled, WfT, bf, obuf, NG, stream);
    split_out_kernel<<<FGC, 256, 0, stream>>>(obuf, (float*)d_out);
}
